// BahdanauAttention_28922309771603
// MI455X (gfx1250) — matrix operand order repaired and checked by reading, not compiled, not run on hardware
//
#include <hip/hip_runtime.h>

#define BB 32
#define TT 1024
#define DD 1024

typedef __attribute__((ext_vector_type(16))) __bf16 v16bf;
typedef __attribute__((ext_vector_type(8)))  float  v8f;
typedef __attribute__((ext_vector_type(4)))  unsigned short v4us;
typedef __attribute__((ext_vector_type(8)))  unsigned short v8us;

__device__ __forceinline__ unsigned short f2bf(float f) {
    unsigned int u = __float_as_uint(f);
    u += 0x7FFFu + ((u >> 16) & 1u);          // round-to-nearest-even
    return (unsigned short)(u >> 16);
}
__device__ __forceinline__ float bf2f(unsigned short s) {
    return __uint_as_float(((unsigned int)s) << 16);
}

// Async DMA: global -> LDS, 16B per lane, tracked by ASYNCcnt (ISA 15.18.3 op 98).
// VDST VGPR carries the per-lane LDS byte offset, VADDR the 64-bit global address.
__device__ __forceinline__ void async_load_b128(void* lds_dst, const void* gsrc) {
#if defined(__gfx1250__)
    unsigned off = (unsigned)(unsigned long long)(__attribute__((address_space(3))) void*)lds_dst;
    asm volatile("global_load_async_to_lds_b128 %0, %1, off"
                 :: "v"(off), "v"(gsrc)
                 : "memory");
#endif
}

// ---------------------------------------------------------------------------
// Prep 1: Wt_hi/Wt_lo[e][d] = bf16 split of W_h[d][e]  (transposed so the
// GEMM B-panel is contiguous along K). LDS 32x33 tile keeps both sides coalesced.
__global__ void prep_w_kernel(const float* __restrict__ Wh,
                              unsigned short* __restrict__ Wt_hi,
                              unsigned short* __restrict__ Wt_lo) {
    __shared__ float tile[32][33];
    const int tx = threadIdx.x, ty = threadIdx.y;
    const int e0 = blockIdx.x * 32, d0 = blockIdx.y * 32;
#pragma unroll
    for (int yy = 0; yy < 4; ++yy) {
        int dl = ty + yy * 8;
        tile[dl][tx] = Wh[(size_t)(d0 + dl) * DD + e0 + tx];
    }
    __syncthreads();
#pragma unroll
    for (int yy = 0; yy < 4; ++yy) {
        int el = ty + yy * 8;
        float v = tile[tx][el];
        unsigned short hi = f2bf(v);
        Wt_hi[(size_t)(e0 + el) * DD + d0 + tx] = hi;
        Wt_lo[(size_t)(e0 + el) * DD + d0 + tx] = f2bf(v - bf2f(hi));
    }
}

// ---------------------------------------------------------------------------
// Prep 2: dec[b][e] = s_t[b,:] @ W_s[:,e] + b_s[e]   (tiny: 67 MFLOP, VALU)
__global__ void prep_dec_kernel(const float* __restrict__ s_t,
                                const float* __restrict__ Ws,
                                const float* __restrict__ b_s,
                                float* __restrict__ dec) {
    const int e = blockIdx.x * 128 + threadIdx.x;
    const int b = blockIdx.y;
    float acc = 0.f;
    for (int d = 0; d < DD; ++d)
        acc = fmaf(s_t[b * DD + d], Ws[(size_t)d * DD + e], acc);
    dec[b * DD + e] = acc + b_s[e];
}

// ---------------------------------------------------------------------------
// Main fused kernel: e_t[m] += V_w . tanh( (h@W_h)[m,:] + dec[b,:] + cov[m]*W_c )
// 64x64 macro-tile per WG, K=1024 accumulated in registers, bf16 hi/lo split
// (3 x v_wmma_f32_16x16x32_bf16 per 16x16x32 step). B panel is double-buffered
// via global_load_async_to_lds_b128 (ASYNCcnt) so the DMA for chunk kc+1
// overlaps WMMA on chunk kc.
__global__ __launch_bounds__(256)
void logits_kernel(const float* __restrict__ h,
                   const unsigned short* __restrict__ Wt_hi,
                   const unsigned short* __restrict__ Wt_lo,
                   const float* __restrict__ dec,
                   const float* __restrict__ coverage,
                   const float* __restrict__ W_c,
                   const float* __restrict__ V_w,
                   float* __restrict__ e_t) {
    constexpr int LDA = 72;                    // halfword stride, bank-conflict-free
    __shared__ unsigned short As_hi[64 * LDA];
    __shared__ unsigned short As_lo[64 * LDA];
    __shared__ unsigned short Bs_hi[2][64 * LDA];
    __shared__ unsigned short Bs_lo[2][64 * LDA];

    const int tid  = threadIdx.x;
    const int lane = tid & 31;
    const int wave = tid >> 5;
    const int ll   = lane & 15;                // lane-in-half
    const int hh   = lane >> 4;                // lane half (WMMA layout)
    const int i_nt = wave & 3;                 // N 16-tile within macro-tile
    const int j_mp = wave >> 2;                // M tile-pair (0,1)

    const int n_base = blockIdx.x * 64;
    const int m_base = blockIdx.y * 64;        // 64 | T so one batch per WG
    const int b      = m_base >> 10;

    v8f acc0 = {};
    v8f acc1 = {};

    // staging thread mapping
    const int ar = tid >> 4;                   // A: row group, col = (tid&15)*4
    const int ac = (tid & 15) * 4;
    const int br = tid >> 2;                   // B: row, col = (tid&3)*16
    const int bc = (tid & 3) * 16;

    // pre-issue async B panel for chunk 0 into buffer 0 (4 async insts / wave)
    {
        const size_t gb = (size_t)(n_base + br) * DD + bc;
        async_load_b128(&Bs_hi[0][br * LDA + bc],     Wt_hi + gb);
        async_load_b128(&Bs_hi[0][br * LDA + bc + 8], Wt_hi + gb + 8);
        async_load_b128(&Bs_lo[0][br * LDA + bc],     Wt_lo + gb);
        async_load_b128(&Bs_lo[0][br * LDA + bc + 8], Wt_lo + gb + 8);
    }

    for (int kc = 0; kc < 16; ++kc) {
        const int k0  = kc * 64;
        const int cur = kc & 1;
        const int nxt = cur ^ 1;
        __syncthreads();                       // previous compute done (A + B[nxt] reusable)
        // ---- stage A (f32 -> bf16 hi/lo) ----
#pragma unroll
        for (int it = 0; it < 4; ++it) {
            const int r = it * 16 + ar;
            const float4 v = *(const float4*)(h + (size_t)(m_base + r) * DD + k0 + ac);
            unsigned short x0 = f2bf(v.x), x1 = f2bf(v.y), x2 = f2bf(v.z), x3 = f2bf(v.w);
            v4us hv = {x0, x1, x2, x3};
            v4us lv = {f2bf(v.x - bf2f(x0)), f2bf(v.y - bf2f(x1)),
                       f2bf(v.z - bf2f(x2)), f2bf(v.w - bf2f(x3))};
            *(v4us*)(&As_hi[r * LDA + ac]) = hv;
            *(v4us*)(&As_lo[r * LDA + ac]) = lv;
        }
        // prefetch next A chunk while this one is consumed
        if (kc + 1 < 16) {
#pragma unroll
            for (int it = 0; it < 4; ++it)
                __builtin_prefetch(h + (size_t)(m_base + it * 16 + ar) * DD + k0 + 64 + ac, 0, 3);
            // ---- kick off async B panel for chunk kc+1 into the other buffer ----
            const size_t gb = (size_t)(n_base + br) * DD + (k0 + 64) + bc;
            async_load_b128(&Bs_hi[nxt][br * LDA + bc],     Wt_hi + gb);
            async_load_b128(&Bs_hi[nxt][br * LDA + bc + 8], Wt_hi + gb + 8);
            async_load_b128(&Bs_lo[nxt][br * LDA + bc],     Wt_lo + gb);
            async_load_b128(&Bs_lo[nxt][br * LDA + bc + 8], Wt_lo + gb + 8);
            // allow the 4 just-issued to stay in flight; drain the 4 for buffer `cur`
            asm volatile("s_wait_asynccnt 0x4" ::: "memory");
        } else {
            asm volatile("s_wait_asynccnt 0x0" ::: "memory");
        }
        __syncthreads();                       // publish A stores + async B arrivals

        // ---- compute: 2 k-steps of 32 ----
#pragma unroll
        for (int s = 0; s < 2; ++s) {
            const int ks = s * 32;
            // B fragment (32x16): lane column n = ll, element e -> k = ks + e + 16*hh
            const int nrow = i_nt * 16 + ll;
            v8us b0 = *(const v8us*)(&Bs_hi[cur][nrow * LDA + ks + 16 * hh]);
            v8us b1 = *(const v8us*)(&Bs_hi[cur][nrow * LDA + ks + 16 * hh + 8]);
            v8us c0 = *(const v8us*)(&Bs_lo[cur][nrow * LDA + ks + 16 * hh]);
            v8us c1 = *(const v8us*)(&Bs_lo[cur][nrow * LDA + ks + 16 * hh + 8]);
            v16bf Bh = __builtin_bit_cast(v16bf,
                __builtin_shufflevector(b0, b1, 0,1,2,3,4,5,6,7,8,9,10,11,12,13,14,15));
            v16bf Bl = __builtin_bit_cast(v16bf,
                __builtin_shufflevector(c0, c1, 0,1,2,3,4,5,6,7,8,9,10,11,12,13,14,15));

            // A fragments (16x32): row m = ll; element e -> k = ks + e + 16*hh
            const int m0r = (j_mp * 2 + 0) * 16 + ll;
            const int m1r = (j_mp * 2 + 1) * 16 + ll;
            const int oa0 = ks + 16 * hh;
            const int oa1 = ks + 16 * hh + 8;
            v8us a00 = *(const v8us*)(&As_hi[m0r * LDA + oa0]);
            v8us a01 = *(const v8us*)(&As_hi[m0r * LDA + oa1]);
            v8us l00 = *(const v8us*)(&As_lo[m0r * LDA + oa0]);
            v8us l01 = *(const v8us*)(&As_lo[m0r * LDA + oa1]);
            v8us a10 = *(const v8us*)(&As_hi[m1r * LDA + oa0]);
            v8us a11 = *(const v8us*)(&As_hi[m1r * LDA + oa1]);
            v8us l10 = *(const v8us*)(&As_lo[m1r * LDA + oa0]);
            v8us l11 = *(const v8us*)(&As_lo[m1r * LDA + oa1]);
            v16bf A0h = __builtin_bit_cast(v16bf,
                __builtin_shufflevector(a00, a01, 0,1,2,3,4,5,6,7,8,9,10,11,12,13,14,15));
            v16bf A0l = __builtin_bit_cast(v16bf,
                __builtin_shufflevector(l00, l01, 0,1,2,3,4,5,6,7,8,9,10,11,12,13,14,15));
            v16bf A1h = __builtin_bit_cast(v16bf,
                __builtin_shufflevector(a10, a11, 0,1,2,3,4,5,6,7,8,9,10,11,12,13,14,15));
            v16bf A1l = __builtin_bit_cast(v16bf,
                __builtin_shufflevector(l10, l11, 0,1,2,3,4,5,6,7,8,9,10,11,12,13,14,15));

            // hi*hi + lo*hi + hi*lo  (f32 accumulate)
            acc0 = __builtin_amdgcn_wmma_f32_16x16x32_bf16(false, A0h, false, Bh, (short)0, acc0, false, false);
            acc0 = __builtin_amdgcn_wmma_f32_16x16x32_bf16(false, A0l, false, Bh, (short)0, acc0, false, false);
            acc0 = __builtin_amdgcn_wmma_f32_16x16x32_bf16(false, A0h, false, Bl, (short)0, acc0, false, false);
            acc1 = __builtin_amdgcn_wmma_f32_16x16x32_bf16(false, A1h, false, Bh, (short)0, acc1, false, false);
            acc1 = __builtin_amdgcn_wmma_f32_16x16x32_bf16(false, A1l, false, Bh, (short)0, acc1, false, false);
            acc1 = __builtin_amdgcn_wmma_f32_16x16x32_bf16(false, A1h, false, Bl, (short)0, acc1, false, false);
        }
    }

    // ---- fused epilogue: +dec +cov*W_c, tanh, .V_w, reduce over N-16, atomics ----
    const int n_g    = n_base + i_nt * 16 + ll;    // per-lane column
    const float decl = dec[b * DD + n_g];
    const float wc   = W_c[n_g];
    const float vw   = V_w[n_g];

#pragma unroll
    for (int p = 0; p < 2; ++p) {
        const int mt = j_mp * 2 + p;
#pragma unroll
        for (int r = 0; r < 8; ++r) {
            const int m_g = m_base + mt * 16 + r + 8 * hh;   // C layout: row = r + 8*half
            const float cov = coverage[m_g];
            const float a = (p == 0) ? acc0[r] : acc1[r];
            float v = tanhf(a + decl + cov * wc) * vw;
            v += __shfl_xor(v, 1, 32);
            v += __shfl_xor(v, 2, 32);
            v += __shfl_xor(v, 4, 32);
            v += __shfl_xor(v, 8, 32);                       // sum over the 16 columns
            if (ll == 0) atomicAdd(&e_t[m_g], v);            // global_atomic_add_f32
        }
    }
}

// ---------------------------------------------------------------------------
// Softmax over T per batch + coverage_new
__global__ __launch_bounds__(256)
void softmax_kernel(const float* __restrict__ e_t,
                    const float* __restrict__ coverage,
                    float* __restrict__ a_out,
                    float* __restrict__ cov_out) {
    __shared__ float red[8];
    const int b = blockIdx.x, tid = threadIdx.x;
    const float4 e4 = ((const float4*)(e_t + b * TT))[tid];
    float m = fmaxf(fmaxf(e4.x, e4.y), fmaxf(e4.z, e4.w));
    for (int o = 1; o < 32; o <<= 1) m = fmaxf(m, __shfl_xor(m, o, 32));
    if ((tid & 31) == 0) red[tid >> 5] = m;
    __syncthreads();
    m = red[0];
#pragma unroll
    for (int w = 1; w < 8; ++w) m = fmaxf(m, red[w]);
    __syncthreads();
    float4 x;
    x.x = __expf(e4.x - m); x.y = __expf(e4.y - m);
    x.z = __expf(e4.z - m); x.w = __expf(e4.w - m);
    float s = x.x + x.y + x.z + x.w;
    for (int o = 1; o < 32; o <<= 1) s += __shfl_xor(s, o, 32);
    if ((tid & 31) == 0) red[tid >> 5] = s;
    __syncthreads();
    s = 0.f;
#pragma unroll
    for (int w = 0; w < 8; ++w) s += red[w];
    const float inv = 1.f / s;
    x.x *= inv; x.y *= inv; x.z *= inv; x.w *= inv;
    ((float4*)(a_out + b * TT))[tid] = x;
    float4 c4 = ((const float4*)(coverage + b * TT))[tid];
    c4.x += x.x; c4.y += x.y; c4.z += x.z; c4.w += x.w;
    ((float4*)(cov_out + b * TT))[tid] = c4;
}

// ---------------------------------------------------------------------------
// context[b][d] = sum_t a_t[b,t] * h_i[b,t,d]   (memory bound; h_i L2-resident)
__global__ __launch_bounds__(256)
void context_kernel(const float* __restrict__ h,
                    const float* __restrict__ a_t,
                    float* __restrict__ ctx) {
    const int b = blockIdx.x;
    const int d = blockIdx.y * 256 + threadIdx.x;
    const float* hb = h + (size_t)b * TT * DD;
    const float* ab = a_t + b * TT;
    float acc = 0.f;
    for (int t = 0; t < TT; ++t)
        acc = fmaf(ab[t], hb[(size_t)t * DD + d], acc);
    ctx[b * DD + d] = acc;
}

// ---------------------------------------------------------------------------
extern "C" void kernel_launch(void* const* d_in, const int* in_sizes, int n_in,
                              void* d_out, int out_size, void* d_ws, size_t ws_size,
                              hipStream_t stream) {
    const float* h_i      = (const float*)d_in[0];
    const float* s_t      = (const float*)d_in[1];
    const float* coverage = (const float*)d_in[2];
    const float* W_h      = (const float*)d_in[3];
    const float* W_s      = (const float*)d_in[4];
    const float* b_s      = (const float*)d_in[5];
    const float* W_c      = (const float*)d_in[6];
    const float* V_w      = (const float*)d_in[7];

    // workspace: Wt_hi (2MB) | Wt_lo (2MB) | dec (128KB) | e_t (128KB)
    unsigned short* Wt_hi = (unsigned short*)d_ws;
    unsigned short* Wt_lo = Wt_hi + (size_t)DD * DD;
    float* dec = (float*)(Wt_lo + (size_t)DD * DD);
    float* e_t = dec + (size_t)BB * DD;

    float* out     = (float*)d_out;
    float* ctx_out = out;                    // (B, D)
    float* a_out   = out + (size_t)BB * DD;  // (B, T)
    float* cov_out = a_out + (size_t)BB * TT;// (B, T)

    hipMemsetAsync(e_t, 0, (size_t)BB * TT * sizeof(float), stream);
    prep_w_kernel<<<dim3(32, 32), dim3(32, 8), 0, stream>>>(W_h, Wt_hi, Wt_lo);
    prep_dec_kernel<<<dim3(8, 32), dim3(128), 0, stream>>>(s_t, W_s, b_s, dec);
    logits_kernel<<<dim3(16, 512), dim3(256), 0, stream>>>(h_i, Wt_hi, Wt_lo, dec,
                                                           coverage, W_c, V_w, e_t);
    softmax_kernel<<<dim3(32), dim3(256), 0, stream>>>(e_t, coverage, a_out, cov_out);
    context_kernel<<<dim3(32, 4), dim3(256), 0, stream>>>(h_i, a_out, ctx_out);
}
